// GRUEncoder_65343632441410
// MI455X (gfx1250) — matrix operand order repaired and checked by reading, not compiled, not run on hardware
//
#include <hip/hip_runtime.h>
#include <cstdint>
#include <cstddef>

// ---------------------------------------------------------------------------
// MI455X (gfx1250) GRU-encoder pipeline.
// All GEMMs (~190 GFLOP total) run through v_wmma_f32_16x16x32_bf16:
// bf16 operands, f32 accumulation. One wave32 computes a 32x64 C tile
// (8 WMMAs per K=32 step, 12 b128 loads -> 1.5 VMEM inst per WMMA).
// Weights stay in (Nout x K) row-major form so both A and B WMMA fragments
// are contiguous 16B loads (B lane n holds row n of W == column n of B).
// ---------------------------------------------------------------------------

typedef __attribute__((ext_vector_type(16))) __bf16      v16bf;
typedef __attribute__((ext_vector_type(8)))  float       v8f;
typedef __attribute__((ext_vector_type(4)))  unsigned int v4u;

union FragBF { v16bf v; v4u u[2]; };

__device__ __forceinline__ float elu1(float x) { return x > 0.f ? x : (expf(x) - 1.f); }
__device__ __forceinline__ float sigm(float x) { return 1.f / (1.f + expf(-x)); }

// ---------------- WMMA GEMM:  C[MxN] = act(A[MxKp] @ W[NxKp]^T + bias) ------
// A, W bf16 row-major; C f32; Kp % 32 == 0, M % 32 == 0, N % 64 == 0.
// blockDim = 256 (8 waves); each wave owns one 32x64 tile.
__device__ __forceinline__ void wmma_epilogue(float* __restrict__ C, v8f acc,
                                              int m0, int n, int N, float bv,
                                              int act, int half) {
#pragma unroll
  for (int r = 0; r < 8; ++r) {
    float v = acc[r] + bv;
    if (act) v = elu1(v);
    C[(size_t)(m0 + half * 8 + r) * N + n] = v;
  }
}

__global__ void gemm_bf16_wmma(const __bf16* __restrict__ A,
                               const __bf16* __restrict__ W,
                               const float* __restrict__ bias,
                               float* __restrict__ C,
                               int M, int N, int Kp, int act) {
  const int lane  = threadIdx.x & 31;
  const int wave  = threadIdx.x >> 5;
  const int tilesN = N >> 6;
  const int tile  = blockIdx.x * 8 + wave;
  const int mt    = tile / tilesN;
  const int nt    = tile - mt * tilesN;
  if (mt * 32 >= M) return;

  const int half = lane >> 4;   // which half-wave
  const int l15  = lane & 15;

  const __bf16* __restrict__ Arow0 = A + (size_t)(mt * 32 + l15) * Kp;
  const __bf16* __restrict__ Arow1 = Arow0 + (size_t)16 * Kp;
  const __bf16* __restrict__ Wbase = W + (size_t)(nt * 64 + l15) * Kp;
  const size_t wstride = (size_t)16 * Kp;   // 16 rows of W per n-subtile

  v8f acc00 = {}, acc01 = {}, acc02 = {}, acc03 = {};
  v8f acc10 = {}, acc11 = {}, acc12 = {}, acc13 = {};

  for (int k0 = 0; k0 < Kp; k0 += 32) {
    // A 16x32 bf16 fragment: half 0 -> K 0..15 ; half 1 -> K 16..31
    FragBF a0, a1;
    a0.u[0] = *(const v4u*)(Arow0 + k0 + half * 16);
    a0.u[1] = *(const v4u*)(Arow0 + k0 + 8 + half * 16);
    a1.u[0] = *(const v4u*)(Arow1 + k0 + half * 16);
    a1.u[1] = *(const v4u*)(Arow1 + k0 + 8 + half * 16);
    // B 32x16 bf16 fragment: lanes 0..15 hold K 0..15, lanes 16..31 hold K 16..31
    const __bf16* wp = Wbase + k0 + half * 16;
    FragBF b0, b1, b2, b3;
    b0.u[0] = *(const v4u*)(wp);                 b0.u[1] = *(const v4u*)(wp + 8);
    b1.u[0] = *(const v4u*)(wp + wstride);       b1.u[1] = *(const v4u*)(wp + wstride + 8);
    b2.u[0] = *(const v4u*)(wp + 2 * wstride);   b2.u[1] = *(const v4u*)(wp + 2 * wstride + 8);
    b3.u[0] = *(const v4u*)(wp + 3 * wstride);   b3.u[1] = *(const v4u*)(wp + 3 * wstride + 8);

    acc00 = __builtin_amdgcn_wmma_f32_16x16x32_bf16(false, a0.v, false, b0.v, (short)0, acc00, false, false);
    acc01 = __builtin_amdgcn_wmma_f32_16x16x32_bf16(false, a0.v, false, b1.v, (short)0, acc01, false, false);
    acc02 = __builtin_amdgcn_wmma_f32_16x16x32_bf16(false, a0.v, false, b2.v, (short)0, acc02, false, false);
    acc03 = __builtin_amdgcn_wmma_f32_16x16x32_bf16(false, a0.v, false, b3.v, (short)0, acc03, false, false);
    acc10 = __builtin_amdgcn_wmma_f32_16x16x32_bf16(false, a1.v, false, b0.v, (short)0, acc10, false, false);
    acc11 = __builtin_amdgcn_wmma_f32_16x16x32_bf16(false, a1.v, false, b1.v, (short)0, acc11, false, false);
    acc12 = __builtin_amdgcn_wmma_f32_16x16x32_bf16(false, a1.v, false, b2.v, (short)0, acc12, false, false);
    acc13 = __builtin_amdgcn_wmma_f32_16x16x32_bf16(false, a1.v, false, b3.v, (short)0, acc13, false, false);
  }

  const int m0 = mt * 32;
  const int n0 = nt * 64 + l15;
  wmma_epilogue(C, acc00, m0,      n0,      N, bias[n0],      act, half);
  wmma_epilogue(C, acc01, m0,      n0 + 16, N, bias[n0 + 16], act, half);
  wmma_epilogue(C, acc02, m0,      n0 + 32, N, bias[n0 + 32], act, half);
  wmma_epilogue(C, acc03, m0,      n0 + 48, N, bias[n0 + 48], act, half);
  wmma_epilogue(C, acc10, m0 + 16, n0,      N, bias[n0],      act, half);
  wmma_epilogue(C, acc11, m0 + 16, n0 + 16, N, bias[n0 + 16], act, half);
  wmma_epilogue(C, acc12, m0 + 16, n0 + 32, N, bias[n0 + 32], act, half);
  wmma_epilogue(C, acc13, m0 + 16, n0 + 48, N, bias[n0 + 48], act, half);
}

// ---------------- f32 -> bf16 pack with K zero-padding ----------------------
__global__ void pack_bf16(const float* __restrict__ src, __bf16* __restrict__ dst,
                          int M, int K, int Kp) {
  size_t total = (size_t)M * Kp;
  size_t i = (size_t)blockIdx.x * blockDim.x + threadIdx.x;
  if (i >= total) return;
  int k = (int)(i % Kp);
  size_t m = i / Kp;
  dst[i] = (k < K) ? (__bf16)src[m * K + k] : (__bf16)0.0f;
}

__global__ void zero_u32(unsigned int* __restrict__ p, size_t n) {
  size_t i = (size_t)blockIdx.x * blockDim.x + threadIdx.x;
  if (i < n) p[i] = 0u;
}

// ---------------- BatchNorm over flattened rows (256 features) --------------
__global__ void bn_stats(const float* __restrict__ X, float* __restrict__ mv, int M) {
  const int f = blockIdx.x;          // one block per feature, 256 threads
  float s = 0.f, ss = 0.f;
  for (int m = threadIdx.x; m < M; m += 256) {
    float v = X[(size_t)m * 256 + f];
    s += v; ss += v * v;
  }
  __shared__ float sh[512];
  sh[threadIdx.x] = s; sh[256 + threadIdx.x] = ss;
  __syncthreads();
  for (int st = 128; st > 0; st >>= 1) {
    if (threadIdx.x < st) {
      sh[threadIdx.x] += sh[threadIdx.x + st];
      sh[256 + threadIdx.x] += sh[256 + threadIdx.x + st];
    }
    __syncthreads();
  }
  if (threadIdx.x == 0) {
    float mean = sh[0] / (float)M;
    float var  = sh[256] / (float)M - mean * mean;
    mv[f] = mean;
    mv[256 + f] = rsqrtf(var + 1e-5f);
  }
}

__global__ void bn_apply(const float* __restrict__ X, const float* __restrict__ mv,
                         const float* __restrict__ gamma, const float* __restrict__ beta,
                         float* __restrict__ Y, size_t total) {
  size_t i = (size_t)blockIdx.x * blockDim.x + threadIdx.x;
  if (i >= total) return;
  int c = (int)(i & 255);
  Y[i] = (X[i] - mv[c]) * mv[256 + c] * gamma[c] + beta[c];
}

// ---------------- one-hot "matmuls" as gathers ------------------------------
// mlp2 input: row (b,e) = [x1[b, e%32] , x1[b, e/32]]   (65536 x 512)
__global__ void gather_edge2(const float* __restrict__ x1, __bf16* __restrict__ dst) {
  size_t i = (size_t)blockIdx.x * blockDim.x + threadIdx.x;
  if (i >= (size_t)65536 * 512) return;
  int c = (int)(i & 511);
  size_t r = i >> 9;
  int b = (int)(r >> 10), e = (int)(r & 1023);
  int node = (c < 256) ? (e & 31) : (e >> 5);
  dst[i] = (__bf16)x1[((size_t)b * 32 + node) * 256 + (c & 255)];
}

// mlp4 input: [x3[b, e%32], x3[b, e/32], xskip[b,e]]  (65536 x 768)
__global__ void gather_edge3(const float* __restrict__ x3, const float* __restrict__ xskip,
                             __bf16* __restrict__ dst) {
  size_t i = (size_t)blockIdx.x * blockDim.x + threadIdx.x;
  if (i >= (size_t)65536 * 768) return;
  int c = (int)(i % 768);
  size_t r = i / 768;
  int b = (int)(r >> 10), e = (int)(r & 1023);
  float v;
  if (c < 512) {
    int node = (c < 256) ? (e & 31) : (e >> 5);
    v = x3[((size_t)b * 32 + node) * 256 + (c & 255)];
  } else {
    v = xskip[r * 256 + (c - 512)];
  }
  dst[i] = (__bf16)v;
}

// x_agg[b,n] = mean_j xskip[b, n*32+j]  -> bf16 (2048 x 256)
__global__ void agg_mean(const float* __restrict__ x2, __bf16* __restrict__ out) {
  int i = blockIdx.x * blockDim.x + threadIdx.x;
  if (i >= 2048 * 256) return;
  int c = i & 255, row = i >> 8;
  int b = row >> 5, n = row & 31;
  const float* p = x2 + ((size_t)b * 1024 + n * 32) * 256 + c;
  float s = 0.f;
#pragma unroll 8
  for (int j = 0; j < 32; ++j) s += p[(size_t)j * 256];
  out[i] = (__bf16)(s * (1.f / 32.f));
}

// GRU1 input: xs[t][s*64+b] = x4[b, ((s+t)%32)*32 + s]   (32 x 2048 x 256)
__global__ void gather_gru1(const float* __restrict__ x4, __bf16* __restrict__ xs) {
  size_t i = (size_t)blockIdx.x * blockDim.x + threadIdx.x;
  if (i >= (size_t)32 * 2048 * 256) return;
  int c = (int)(i & 255);
  size_t rr = i >> 8;
  int row = (int)(rr & 2047), t = (int)(rr >> 11);
  int s = row >> 6, b = row & 63;
  int node = (s + t) & 31;
  xs[i] = (__bf16)x4[((size_t)b * 1024 + node * 32 + s) * 256 + c];
}

// x_edges[s*64+b] = mean_t ys1[t][s*64+b]
__global__ void edge_mean(const float* __restrict__ ys1, float* __restrict__ xe) {
  int i = blockIdx.x * blockDim.x + threadIdx.x;
  if (i >= 2048 * 256) return;
  float s = 0.f;
#pragma unroll 8
  for (int t = 0; t < 32; ++t) s += ys1[(size_t)t * 524288 + i];
  xe[i] = s * (1.f / 32.f);
}

// GRU2 input: xs2[t][s*64+b] = x_edges[(s+t)%32, b]
__global__ void gather_gru2(const float* __restrict__ xe, __bf16* __restrict__ xs) {
  size_t i = (size_t)blockIdx.x * blockDim.x + threadIdx.x;
  if (i >= (size_t)32 * 2048 * 256) return;
  int c = (int)(i & 255);
  size_t rr = i >> 8;
  int row = (int)(rr & 2047), t = (int)(rr >> 11);
  int s = row >> 6, b = row & 63;
  xs[i] = (__bf16)xe[((size_t)(((s + t) & 31) * 64 + b)) * 256 + c];
}

// fused GRU gates; h updated in place, bf16 copy for next step's WMMA A.
__global__ void gru_gate(const float* __restrict__ gi, const float* __restrict__ gh,
                         float* __restrict__ h, __bf16* __restrict__ hbf,
                         float* __restrict__ ys) {
  int i = blockIdx.x * blockDim.x + threadIdx.x;
  if (i >= 524288) return;
  int m = i >> 8, c = i & 255;
  size_t base = (size_t)m * 768 + c;
  float r = sigm(gi[base]       + gh[base]);
  float z = sigm(gi[base + 256] + gh[base + 256]);
  float n = tanhf(gi[base + 512] + r * gh[base + 512]);
  float hv = (1.f - z) * n + z * h[i];
  h[i] = hv; hbf[i] = (__bf16)hv; ys[i] = hv;
}

// mlp5 input: row (b, s*32+t) = [ys1[t][s*64+b], ys2[t][s*64+b]] (65536 x 512)
__global__ void build_mlp5(const float* __restrict__ ys1, const float* __restrict__ ys2,
                           __bf16* __restrict__ dst) {
  size_t i = (size_t)blockIdx.x * blockDim.x + threadIdx.x;
  if (i >= (size_t)65536 * 512) return;
  int c = (int)(i & 511);
  size_t r = i >> 9;
  int b = (int)(r >> 10), e = (int)(r & 1023);
  int s = e >> 5, t = e & 31;
  size_t grow = ((size_t)s * 64 + b) * 256;
  float v = (c < 256) ? ys1[(size_t)t * 524288 + grow + c]
                      : ys2[(size_t)t * 524288 + grow + (c - 256)];
  dst[i] = (__bf16)v;
}

// head: 256 -> 2
__global__ void fcout_kernel(const float* __restrict__ X, const float* __restrict__ w,
                             const float* __restrict__ b, float* __restrict__ out, int M) {
  int m = blockIdx.x * blockDim.x + threadIdx.x;
  if (m >= M) return;
  float a0 = 0.f, a1 = 0.f;
  const float* x = X + (size_t)m * 256;
#pragma unroll 8
  for (int k = 0; k < 256; ++k) { a0 += x[k] * w[k]; a1 += x[k] * w[256 + k]; }
  out[(size_t)m * 2]     = a0 + b[0];
  out[(size_t)m * 2 + 1] = a1 + b[1];
}

// ---------------------------------------------------------------------------
extern "C" void kernel_launch(void* const* d_in, const int* in_sizes, int n_in,
                              void* d_out, int out_size, void* d_ws, size_t ws_size,
                              hipStream_t stream) {
  (void)in_sizes; (void)n_in; (void)out_size; (void)ws_size;

  const float* inputs = (const float*)d_in[0];
  // params flattened depth-first in dict order: mlp1..mlp5 (w1,b1,w2,b2,g,be),
  // gru1/gru2 (wih,whh,bih,bhh), fc_out (w,b)
  const float* P[40];
  for (int i = 0; i < 40; ++i) P[i] = (const float*)d_in[3 + i];
  const float *m_w1[5], *m_b1[5], *m_w2[5], *m_b2[5], *m_g[5], *m_be[5];
  for (int i = 0; i < 5; ++i) {
    m_w1[i] = P[i * 6 + 0]; m_b1[i] = P[i * 6 + 1]; m_w2[i] = P[i * 6 + 2];
    m_b2[i] = P[i * 6 + 3]; m_g[i]  = P[i * 6 + 4]; m_be[i] = P[i * 6 + 5];
  }
  const float* g_wih[2] = { P[30], P[34] };
  const float* g_whh[2] = { P[31], P[35] };
  const float* g_bih[2] = { P[32], P[36] };
  const float* g_bhh[2] = { P[33], P[37] };
  const float* fc_w = P[38];
  const float* fc_b = P[39];

  // ---- workspace carve-out (deterministic static layout) ----
  char* ws = (char*)d_ws;
  size_t off = 0;
  auto alloc = [&](size_t bytes) -> char* {
    char* p = ws + off; off = (off + bytes + 255) & ~(size_t)255; return p;
  };
  const int mlpK[5]  = { 196, 512, 256, 768, 512 };
  const int mlpKp[5] = { 224, 512, 256, 768, 512 };
  __bf16 *W1p[5], *W2p[5];
  for (int i = 0; i < 5; ++i) {
    W1p[i] = (__bf16*)alloc((size_t)256 * mlpKp[i] * 2);
    W2p[i] = (__bf16*)alloc((size_t)256 * 256 * 2);
  }
  __bf16* GWi[2]; __bf16* GWh[2];
  for (int i = 0; i < 2; ++i) {
    GWi[i] = (__bf16*)alloc((size_t)768 * 256 * 2);
    GWh[i] = (__bf16*)alloc((size_t)768 * 256 * 2);
  }
  __bf16* XIN   = (__bf16*)alloc((size_t)2048 * 224 * 2);
  float*  NF1   = (float*) alloc((size_t)2048 * 256 * 4);
  float*  NF2   = (float*) alloc((size_t)2048 * 256 * 4);
  __bf16* NBF   = (__bf16*)alloc((size_t)2048 * 256 * 2);
  __bf16* AGGBF = (__bf16*)alloc((size_t)2048 * 256 * 2);
  float*  MV    = (float*) alloc(512 * 4);
  float*  BF1   = (float*) alloc((size_t)65536 * 256 * 4);
  float*  BF2   = (float*) alloc((size_t)65536 * 256 * 4);
  float*  XSKIP = (float*) alloc((size_t)65536 * 256 * 4);
  float*  X4    = (float*) alloc((size_t)65536 * 256 * 4);
  __bf16* BIGN  = (__bf16*)alloc((size_t)65536 * 256 * 2);
  __bf16* BIGW  = (__bf16*)alloc((size_t)65536 * 768 * 2);  // widest edge input
  __bf16* XS1   = (__bf16*)alloc((size_t)32 * 2048 * 256 * 2);
  __bf16* XS2   = (__bf16*)alloc((size_t)32 * 2048 * 256 * 2);
  float*  YS1   = (float*) alloc((size_t)32 * 2048 * 256 * 4);
  float*  YS2   = (float*) alloc((size_t)32 * 2048 * 256 * 4);
  float*  GI    = (float*) alloc((size_t)2048 * 768 * 4);
  float*  GH    = (float*) alloc((size_t)2048 * 768 * 4);
  float*  HF    = (float*) alloc((size_t)2048 * 256 * 4);
  __bf16* HBF   = (__bf16*)alloc((size_t)2048 * 256 * 2);
  float*  XE    = (float*) alloc((size_t)2048 * 256 * 4);

  auto nblk = [](size_t total) { return (unsigned)((total + 255) / 256); };
  auto run_pack = [&](const float* src, __bf16* dst, int M, int K, int Kp) {
    pack_bf16<<<nblk((size_t)M * Kp), 256, 0, stream>>>(src, dst, M, K, Kp);
  };
  auto run_gemm = [&](const __bf16* A, const __bf16* Wm, const float* bias, float* C,
                      int M, int N, int Kp, int act) {
    int tiles = (M / 32) * (N / 64);
    gemm_bf16_wmma<<<(tiles + 7) / 8, 256, 0, stream>>>(A, Wm, bias, C, M, N, Kp, act);
  };
  auto run_bn = [&](const float* X, float* Y, const float* gamma, const float* beta, int M) {
    bn_stats<<<256, 256, 0, stream>>>(X, MV, M);
    bn_apply<<<nblk((size_t)M * 256), 256, 0, stream>>>(X, MV, gamma, beta, Y, (size_t)M * 256);
  };
  // MLP: elu(gemm1) -> pack -> elu(gemm2) -> BN  (output f32 in `out`)
  auto run_mlp = [&](int idx, const __bf16* A, int M, float* t1, __bf16* tbf, float* t2,
                     float* out) {
    run_gemm(A, W1p[idx], m_b1[idx], t1, M, 256, mlpKp[idx], 1);
    run_pack(t1, tbf, M, 256, 256);
    run_gemm(tbf, W2p[idx], m_b2[idx], t2, M, 256, 256, 1);
    run_bn(t2, out, m_g[idx], m_be[idx], M);
  };
  auto run_gru = [&](const __bf16* XS, int gi_idx, float* YS) {
    zero_u32<<<nblk(524288), 256, 0, stream>>>((unsigned int*)HF, 524288);
    zero_u32<<<nblk(262144), 256, 0, stream>>>((unsigned int*)HBF, 262144);
    for (int t = 0; t < 32; ++t) {
      run_gemm(XS + (size_t)t * 2048 * 256, GWi[gi_idx], g_bih[gi_idx], GI, 2048, 768, 256, 0);
      run_gemm(HBF, GWh[gi_idx], g_bhh[gi_idx], GH, 2048, 768, 256, 0);
      gru_gate<<<nblk(524288), 256, 0, stream>>>(GI, GH, HF, HBF, YS + (size_t)t * 524288);
    }
  };

  // ---- pack all weights to bf16 (mlp1.w1 zero-padded K 196->224) ----
  for (int i = 0; i < 5; ++i) {
    run_pack(m_w1[i], W1p[i], 256, mlpK[i], mlpKp[i]);
    run_pack(m_w2[i], W2p[i], 256, 256, 256);
  }
  for (int i = 0; i < 2; ++i) {
    run_pack(g_wih[i], GWi[i], 768, 256, 256);
    run_pack(g_whh[i], GWh[i], 768, 256, 256);
  }

  // ---- mlp1 on nodes (2048 x 196) ----
  run_pack(inputs, XIN, 2048, 196, 224);
  run_mlp(0, XIN, 2048, NF2, NBF, NF2, NF1);                 // x1 -> NF1

  // ---- edge concat + mlp2 ----
  gather_edge2<<<nblk((size_t)65536 * 512), 256, 0, stream>>>(NF1, BIGW);
  run_mlp(1, BIGW, 65536, BF1, BIGN, BF2, XSKIP);            // x_skip -> XSKIP

  // ---- receiver mean + mlp3 ----
  agg_mean<<<nblk(2048 * 256), 256, 0, stream>>>(XSKIP, AGGBF);
  run_mlp(2, AGGBF, 2048, NF2, NBF, NF2, NF1);               // x3 -> NF1

  // ---- edge concat3 + mlp4 ----
  gather_edge3<<<nblk((size_t)65536 * 768), 256, 0, stream>>>(NF1, XSKIP, BIGW);
  run_mlp(3, BIGW, 65536, BF1, BIGN, BF2, X4);               // x4 -> X4

  // ---- GRU1 over diagonal reindexing ----
  gather_gru1<<<nblk((size_t)32 * 2048 * 256), 256, 0, stream>>>(X4, XS1);
  run_gru(XS1, 0, YS1);

  // ---- mean over t, GRU2 ----
  edge_mean<<<nblk(2048 * 256), 256, 0, stream>>>(YS1, XE);
  gather_gru2<<<nblk((size_t)32 * 2048 * 256), 256, 0, stream>>>(XE, XS2);
  run_gru(XS2, 1, YS2);

  // ---- mlp5 + head ----
  build_mlp5<<<nblk((size_t)65536 * 512), 256, 0, stream>>>(YS1, YS2, BIGW);
  run_mlp(4, BIGW, 65536, BF1, BIGN, BF2, BF1);              // mlp5 out -> BF1
  fcout_kernel<<<nblk(65536), 256, 0, stream>>>(BF1, fc_w, fc_b, (float*)d_out, 65536);
}
